// GATLayer_44392781971654
// MI455X (gfx1250) — matrix operand order repaired and checked by reading, not compiled, not run on hardware
//
#include <hip/hip_runtime.h>
#include <hip/hip_bf16.h>

typedef __attribute__((ext_vector_type(16))) __bf16 v16bf;
typedef __attribute__((ext_vector_type(8)))  __bf16 v8bf;
typedef __attribute__((ext_vector_type(8)))  float  v8f;
typedef __attribute__((ext_vector_type(4)))  float  v4f;

#define NEG_SLOPE 0.2f
#define RSQRT_FH  0.17677669529663687f   // 1/sqrt(32)

// ---------------------------------------------------------------------------
// Zero-fill (re-init accumulators every call)
// ---------------------------------------------------------------------------
__global__ void zero_f32(float* __restrict__ p, long long n) {
    long long i = (long long)blockIdx.x * blockDim.x + threadIdx.x;
    long long stride = (long long)gridDim.x * blockDim.x;
    for (; i < n; i += stride) p[i] = 0.0f;
}

// ---------------------------------------------------------------------------
// f32 -> bf16 elementwise (for A operands: h, agg)
// ---------------------------------------------------------------------------
__global__ void cvt_bf16(const float* __restrict__ src, __bf16* __restrict__ dst,
                         long long n) {
    long long i = (long long)blockIdx.x * blockDim.x + threadIdx.x;
    long long stride = (long long)gridDim.x * blockDim.x;
    for (; i < n; i += stride) dst[i] = (__bf16)src[i];
}

// ---------------------------------------------------------------------------
// 128x128 weight: f32 row-major (k,n) -> bf16 transposed Wt[n][k]
// ---------------------------------------------------------------------------
__global__ __launch_bounds__(256) void cvt_w_t(const float* __restrict__ W,
                                               __bf16* __restrict__ Wt) {
    const int idx = blockIdx.x * 256 + threadIdx.x;   // 0..16383
    const int n = idx >> 7;
    const int k = idx & 127;
    Wt[idx] = (__bf16)W[k * 128 + n];
}

// ---------------------------------------------------------------------------
// WMMA GEMM: out[M x 128] = act(A_bf16[M x 128] @ W + bias)
// A: bf16 row-major. Wt: bf16 transposed [n][k]. fp32 accumulate.
// Block = 256 threads (8 wave32); block tile = 128x128.
// Wave w: 16-col strip w x 8 row tiles. All addresses hoisted; the K loop is
// pure {global_load_b128, v_wmma} with immediate offsets.
// ---------------------------------------------------------------------------
__global__ __launch_bounds__(256) void wmma_gemm128(
    const __bf16* __restrict__ A, const __bf16* __restrict__ Wt,
    const float* __restrict__ bias, float* __restrict__ out,
    int M, int leaky)
{
    const int lane  = threadIdx.x & 31;
    const int wave  = threadIdx.x >> 5;      // 0..7 -> column tile
    const int col0  = wave << 4;
    const int half  = lane >> 4;             // 0 or 1
    const int n     = lane & 15;
    const int rowBase = blockIdx.x << 7;

    // Hoisted base pointers (clamped rows; guard applied only at store)
    const __bf16* aptr[8];
    #pragma unroll
    for (int rt = 0; rt < 8; ++rt) {
        int m  = rowBase + (rt << 4) + n;
        int mm = (m < M) ? m : (M - 1);
        aptr[rt] = A + (size_t)mm * 128 + (half << 4);
    }
    const __bf16* bptr = Wt + (size_t)(col0 + n) * 128 + (half << 4);

    v8f acc[8] = {};

    #pragma unroll
    for (int kb = 0; kb < 128; kb += 32) {
        // B fragment: 16 contiguous bf16 = 32B at immediate offset kb*2
        const v16bf bfrag = *(const v16bf*)(bptr + kb);

        #pragma unroll
        for (int rt = 0; rt < 8; ++rt) {
            const v8bf c0 = *(const v8bf*)(aptr[rt] + kb);        // k=kb+half*16..+7
            const v8bf c1 = *(const v8bf*)(aptr[rt] + kb + 8);    // k=kb+half*16+8..+15
            v16bf afrag;
            #pragma unroll
            for (int i = 0; i < 8; ++i) { afrag[i] = c0[i]; afrag[i + 8] = c1[i]; }

            acc[rt] = __builtin_amdgcn_wmma_f32_16x16x32_bf16(
                false, afrag, false, bfrag, (short)0, acc[rt], false, false);
        }
    }

    const float bv = bias[col0 + n];
    float* obase = out + (size_t)rowBase * 128 + col0 + n + (size_t)(half << 3) * 128;

    if (rowBase + 128 <= M) {
        // Fast path: all 128 rows in range, branchless stores
        #pragma unroll
        for (int rt = 0; rt < 8; ++rt) {
            #pragma unroll
            for (int v = 0; v < 8; ++v) {
                float x = acc[rt][v] + bv;
                if (leaky) x = (x > 0.0f) ? x : NEG_SLOPE * x;
                obase[(size_t)((rt << 4) + v) * 128] = x;
            }
        }
    } else {
        #pragma unroll
        for (int rt = 0; rt < 8; ++rt) {
            #pragma unroll
            for (int v = 0; v < 8; ++v) {
                const int row = rowBase + (rt << 4) + v + (half << 3);
                if (row < M) {
                    float x = acc[rt][v] + bv;
                    if (leaky) x = (x > 0.0f) ? x : NEG_SLOPE * x;
                    out[(size_t)row * 128 + col0 + n] = x;
                }
            }
        }
    }
}

// ---------------------------------------------------------------------------
// Edge attention: one wave32 per edge. Lane l owns dims 4l..4l+3 (head l>>3):
// one b128 gather from Q[src] and K[tgt], 3-step shfl_xor reduce within the
// 8-lane head group, then one lane per head does bias+exp+deg atomic.
// ---------------------------------------------------------------------------
__global__ __launch_bounds__(256) void edge_attn(
    const float* __restrict__ Q, const float* __restrict__ K,
    const int* __restrict__ ei, const float* __restrict__ ew,
    const float* __restrict__ We, const float* __restrict__ be,
    float* __restrict__ attn_exp, float* __restrict__ deg, int E)
{
    const int e    = (int)(((long long)blockIdx.x * blockDim.x + threadIdx.x) >> 5);
    const int lane = threadIdx.x & 31;
    if (e >= E) return;

    const int src = ei[e];
    const int tgt = ei[E + e];
    const v4f q = *(const v4f*)(Q + (size_t)src * 128 + lane * 4);
    const v4f k = *(const v4f*)(K + (size_t)tgt * 128 + lane * 4);

    float p = q.x * k.x + q.y * k.y + q.z * k.z + q.w * k.w;
    p += __shfl_xor(p, 1, 32);
    p += __shfl_xor(p, 2, 32);
    p += __shfl_xor(p, 4, 32);      // all 8 lanes of a head group hold the dot

    if ((lane & 7) == 0) {
        const int hd = lane >> 3;
        float eb = ew[e] * We[hd] + be[hd];
        eb = (eb > 0.0f) ? eb : NEG_SLOPE * eb;
        const float a = __expf(p * RSQRT_FH + eb);
        attn_exp[(size_t)e * 4 + hd] = a;
        atomicAdd(&deg[(size_t)tgt * 4 + hd], a);
    }
}

// ---------------------------------------------------------------------------
// Message scatter: one wave32 per edge; lane l owns dims 4l..4l+3 (head l>>3).
// One b128 V gather + 4 f32 atomics into the L2-resident agg table.
// ---------------------------------------------------------------------------
__global__ __launch_bounds__(256) void edge_msg(
    const float* __restrict__ Vv, const int* __restrict__ ei,
    const float* __restrict__ ew, const float* __restrict__ attn_exp,
    const float* __restrict__ deg, float* __restrict__ agg, int E)
{
    const int e    = (int)(((long long)blockIdx.x * blockDim.x + threadIdx.x) >> 5);
    const int lane = threadIdx.x & 31;
    if (e >= E) return;

    const int src = ei[e];
    const int tgt = ei[E + e];
    const int hd  = lane >> 3;

    const float c = attn_exp[(size_t)e * 4 + hd]
                    / (deg[(size_t)tgt * 4 + hd] + 1e-16f) * ew[e];

    const v4f v = *(const v4f*)(Vv + (size_t)src * 128 + lane * 4);
    float* at = agg + (size_t)tgt * 128 + lane * 4;
    atomicAdd(at + 0, v.x * c);
    atomicAdd(at + 1, v.y * c);
    atomicAdd(at + 2, v.z * c);
    atomicAdd(at + 3, v.w * c);
}

// ---------------------------------------------------------------------------
// Host launcher
// ---------------------------------------------------------------------------
extern "C" void kernel_launch(void* const* d_in, const int* in_sizes, int n_in,
                              void* d_out, int out_size, void* d_ws, size_t ws_size,
                              hipStream_t stream) {
    const float* h   = (const float*)d_in[0];
    const int*   ei  = (const int*)  d_in[1];
    const float* ew  = (const float*)d_in[2];
    const float* Wq  = (const float*)d_in[3];
    const float* bq  = (const float*)d_in[4];
    const float* Wk  = (const float*)d_in[5];
    const float* bk  = (const float*)d_in[6];
    const float* Wv  = (const float*)d_in[7];
    const float* bv  = (const float*)d_in[8];
    const float* Wo  = (const float*)d_in[9];
    const float* bo  = (const float*)d_in[10];
    const float* We  = (const float*)d_in[11];
    const float* be  = (const float*)d_in[12];
    float*       out = (float*)d_out;

    const int V = in_sizes[0] / 128;
    const int E = in_sizes[2];
    const size_t VD = (size_t)V * 128;

    // Workspace layout (floats):
    //   Q | K | Vv | agg | deg | attn_exp | hbf(bf16) | aggbf(bf16) | Wt[4](bf16)
    float* Q    = (float*)d_ws;
    float* Kp   = Q    + VD;
    float* Vp   = Kp   + VD;
    float* agg  = Vp   + VD;
    float* deg  = agg  + VD;
    float* attn = deg  + (size_t)V * 4;
    __bf16* hbf   = (__bf16*)(attn + (size_t)E * 4);
    __bf16* aggbf = hbf + VD;
    __bf16* WtQ   = aggbf + VD;
    __bf16* WtK   = WtQ + 128 * 128;
    __bf16* WtV   = WtK + 128 * 128;
    __bf16* WtO   = WtV + 128 * 128;

    // Zero accumulators (agg + deg contiguous region)
    {
        const long long nz = (long long)VD + (long long)V * 4;
        int blocks = (int)((nz + 255) / 256);
        if (blocks > 4096) blocks = 4096;
        zero_f32<<<blocks, 256, 0, stream>>>(agg, nz);
    }

    // Operand pre-conversion: h -> bf16, weights -> bf16 transposed
    {
        int blocks = (int)((VD + 255) / 256);
        if (blocks > 4096) blocks = 4096;
        cvt_bf16<<<blocks, 256, 0, stream>>>(h, hbf, (long long)VD);
        cvt_w_t<<<64, 256, 0, stream>>>(Wq, WtQ);
        cvt_w_t<<<64, 256, 0, stream>>>(Wk, WtK);
        cvt_w_t<<<64, 256, 0, stream>>>(Wv, WtV);
        cvt_w_t<<<64, 256, 0, stream>>>(Wo, WtO);
    }

    const int gemmBlocks = (V + 127) / 128;
    wmma_gemm128<<<gemmBlocks, 256, 0, stream>>>(hbf, WtQ, bq, Q,  V, 0);
    wmma_gemm128<<<gemmBlocks, 256, 0, stream>>>(hbf, WtK, bk, Kp, V, 0);
    wmma_gemm128<<<gemmBlocks, 256, 0, stream>>>(hbf, WtV, bv, Vp, V, 0);

    const int edgeBlocks = (E + 7) / 8;   // 8 wave32 edges per 256-thread block
    edge_attn<<<edgeBlocks, 256, 0, stream>>>(Q, Kp, ei, ew, We, be, attn, deg, E);
    edge_msg <<<edgeBlocks, 256, 0, stream>>>(Vp, ei, ew, attn, deg, agg, E);

    // agg -> bf16, then output projection with LeakyReLU
    {
        int blocks = (int)((VD + 255) / 256);
        if (blocks > 4096) blocks = 4096;
        cvt_bf16<<<blocks, 256, 0, stream>>>(agg, aggbf, (long long)VD);
    }
    wmma_gemm128<<<gemmBlocks, 256, 0, stream>>>(aggbf, WtO, bo, out, V, 1);
}
